// GNN_69758858822498
// MI455X (gfx1250) — matrix operand order repaired and checked by reading, not compiled, not run on hardware
//
#include <hip/hip_runtime.h>
#include <hip/hip_bf16.h>

typedef __attribute__((ext_vector_type(16))) __bf16 v16bf;
typedef __attribute__((ext_vector_type(8)))  float  v8f;

#define N_NODES 1000000
#define N_EDGES 8000000
#define NGRAPH  100000
#define HID     32
#define FLATD   320

// ---------------- GCN phase ----------------

__global__ void k_init_deg(float* deg) {
    int i = blockIdx.x * blockDim.x + threadIdx.x;
    if (i < N_NODES) deg[i] = 1.0f;   // self-loop
}

__global__ void k_deg(const int* __restrict__ dst, float* deg) {
    int e = blockIdx.x * blockDim.x + threadIdx.x;
    if (e < N_EDGES) unsafeAtomicAdd(&deg[dst[e]], 1.0f);
}

__global__ void k_dinv(float* deg) {
    int i = blockIdx.x * blockDim.x + threadIdx.x;
    if (i < N_NODES) deg[i] = rsqrtf(deg[i]);   // deg >= 1 always
}

// acc[n*32+c] = (x@Wc)[n][c] * dinv[n]^2   (self-loop term)
__global__ void k_self(const float* __restrict__ x, const float* __restrict__ Wc,
                       const float* __restrict__ dinv, float* __restrict__ acc) {
    int id = blockIdx.x * blockDim.x + threadIdx.x;   // N_NODES*32 threads
    int n = id >> 5, c = id & 31;
    float di = dinv[n];
    float h  = x[2 * n] * Wc[c] + x[2 * n + 1] * Wc[32 + c];
    acc[id] = h * di * di;
}

// per (edge, channel): recompute h0[src] from x (8B instead of 128B gather)
__global__ void k_scatter(const int* __restrict__ src, const int* __restrict__ dst,
                          const float* __restrict__ x, const float* __restrict__ Wc,
                          const float* __restrict__ dinv, float* __restrict__ acc) {
    int id = blockIdx.x * blockDim.x + threadIdx.x;   // N_EDGES*32 threads (<2^31)
    int e = id >> 5, c = id & 31;
    int s = src[e], d = dst[e];
    float nrm = dinv[s] * dinv[d];
    float h   = x[2 * s] * Wc[c] + x[2 * s + 1] * Wc[32 + c];
    unsafeAtomicAdd(&acc[d * 32 + c], h * nrm);
}

// ---------------- W1 fragment packing (fp32 -> bf16 hi/lo) ----------------
// B fragment layout (16-bit B, 32x16): lane = half*16 + col(0..15),
// per lane two runs of 8 K values (K = kstep*32 + (e>>3)*16 + half*8 + (e&7)).
__global__ void k_pack_w1(const float* __restrict__ W1,
                          __bf16* __restrict__ Bhi, __bf16* __restrict__ Blo) {
    int idx = blockIdx.x * blockDim.x + threadIdx.x;
    if (idx >= 16 * 10 * 32 * 16) return;            // ((ntile*10+ks)*32+lane)*16+e
    int e    = idx & 15;
    int lane = (idx >> 4) & 31;
    int t    = idx >> 9;
    int ks   = t % 10;
    int nt   = t / 10;
    int k    = ks * 32 + ((e >> 3) << 4) + ((lane >> 4) << 3) + (e & 7);
    int col  = (nt << 4) + (lane & 15);
    float v  = W1[k * 256 + col];
    __bf16 hi = (__bf16)v;
    Bhi[idx] = hi;
    Blo[idx] = (__bf16)(v - (float)hi);
}

// ---------------- GEMM: fea = relu(acc+bc)[100K,320] @ W1 + b1 ----------------
__global__ void __launch_bounds__(256) k_gemm(
        const float* __restrict__ acc, const float* __restrict__ bc,
        const __bf16* __restrict__ Bhi, const __bf16* __restrict__ Blo,
        const float* __restrict__ b1, float* __restrict__ fea) {
    __shared__ __align__(32) __bf16 Ahi[10 * 32 * 16];
    __shared__ __align__(32) __bf16 Alo[10 * 32 * 16];
    int tid = threadIdx.x;
    int r0  = blockIdx.x * 16;

    // Fill A fragments in LDS, fused relu(acc+bc), bf16 hi/lo split.
    // A (16-bit, 16x32) map: lane = half*16 + m, K = base16 + half*8 + 2*(v&3)+pos
    for (int f = tid; f < 16 * FLATD; f += 256) {
        int m = f / FLATD, k = f - m * FLATD;
        float v = fmaxf(acc[(r0 + m) * FLATD + k] + bc[k & 31], 0.0f);
        __bf16 hi = (__bf16)v;
        __bf16 lo = (__bf16)(v - (float)hi);
        int ks   = k >> 5, kk = k & 31;
        int half = (kk >> 3) & 1;
        int vreg = ((kk & 16) >> 2) + ((kk & 7) >> 1);
        int eidx = (vreg << 1) | (kk & 1);
        int p    = (ks * 32 + ((half << 4) | m)) * 16 + eidx;
        Ahi[p] = hi;
        Alo[p] = lo;
    }
    __syncthreads();

    int wave = tid >> 5, lane = tid & 31;
    int nt0 = wave * 2, nt1 = nt0 + 1;
    v8f c0 = {}; v8f c1 = {};
    #pragma unroll
    for (int ks = 0; ks < 10; ++ks) {
        v16bf ahi = *(const v16bf*)&Ahi[(ks * 32 + lane) * 16];
        v16bf alo = *(const v16bf*)&Alo[(ks * 32 + lane) * 16];
        v16bf b0h = *(const v16bf*)&Bhi[((nt0 * 10 + ks) * 32 + lane) * 16];
        v16bf b0l = *(const v16bf*)&Blo[((nt0 * 10 + ks) * 32 + lane) * 16];
        v16bf b1h = *(const v16bf*)&Bhi[((nt1 * 10 + ks) * 32 + lane) * 16];
        v16bf b1l = *(const v16bf*)&Blo[((nt1 * 10 + ks) * 32 + lane) * 16];
        // bf16x3 emulation of fp32: hi*hi + hi*lo + lo*hi (lo*lo negligible)
        c0 = __builtin_amdgcn_wmma_f32_16x16x32_bf16(false, ahi, false, b0h, (short)0, c0, false, false);
        c0 = __builtin_amdgcn_wmma_f32_16x16x32_bf16(false, ahi, false, b0l, (short)0, c0, false, false);
        c0 = __builtin_amdgcn_wmma_f32_16x16x32_bf16(false, alo, false, b0h, (short)0, c0, false, false);
        c1 = __builtin_amdgcn_wmma_f32_16x16x32_bf16(false, ahi, false, b1h, (short)0, c1, false, false);
        c1 = __builtin_amdgcn_wmma_f32_16x16x32_bf16(false, ahi, false, b1l, (short)0, c1, false, false);
        c1 = __builtin_amdgcn_wmma_f32_16x16x32_bf16(false, alo, false, b1h, (short)0, c1, false, false);
    }
    // D layout: VGPR r -> lanes 0-15: (M=r, N=lane); lanes 16-31: (M=8+r, N=lane-16)
    int mb  = (lane >> 4) << 3;
    int nc0 = nt0 * 16 + (lane & 15);
    int nc1 = nt1 * 16 + (lane & 15);
    float bb0 = b1[nc0], bb1 = b1[nc1];
    #pragma unroll
    for (int r = 0; r < 8; ++r) {
        int row = r0 + mb + r;
        fea[row * 256 + nc0] = c0[r] + bb0;
        fea[row * 256 + nc1] = c1[r] + bb1;
    }
}

// ---------------- normalize + sigmoid/softplus heads ----------------
__global__ void k_head(float* __restrict__ fea,
                       const float* __restrict__ W2, const float* __restrict__ b2,
                       const float* __restrict__ W3, const float* __restrict__ b3,
                       float* __restrict__ mu, float* __restrict__ theta) {
    int tid  = threadIdx.x;
    int lane = tid & 31;
    int row  = blockIdx.x * 8 + (tid >> 5);   // one wave32 per row
    float* rp = fea + (size_t)row * 256;
    float v[8];
    float ss = 0.0f;
    #pragma unroll
    for (int j = 0; j < 8; ++j) { v[j] = rp[lane + j * 32]; ss += v[j] * v[j]; }
    #pragma unroll
    for (int off = 16; off; off >>= 1) ss += __shfl_xor(ss, off, 32);
    float inv = 1.0f / fmaxf(sqrtf(ss), 1e-12f);
    float d2 = 0.0f, d3 = 0.0f;
    #pragma unroll
    for (int j = 0; j < 8; ++j) {
        float fn = v[j] * inv;
        rp[lane + j * 32] = fn;
        d2 += fn * W2[lane + j * 32];
        d3 += fn * W3[lane + j * 32];
    }
    #pragma unroll
    for (int off = 16; off; off >>= 1) {
        d2 += __shfl_xor(d2, off, 32);
        d3 += __shfl_xor(d3, off, 32);
    }
    if (lane == 0) {
        float z2 = d2 + b2[0];
        mu[row] = 1.0f / (1.0f + expf(-z2));
        float z3 = d3 + b3[0];
        theta[row] = (z3 > 30.0f) ? z3 : log1pf(expf(z3));
    }
}

// ---------------- launcher ----------------
extern "C" void kernel_launch(void* const* d_in, const int* in_sizes, int n_in,
                              void* d_out, int out_size, void* d_ws, size_t ws_size,
                              hipStream_t stream) {
    (void)in_sizes; (void)n_in; (void)out_size; (void)ws_size;
    const float* x  = (const float*)d_in[0];
    const int*   ei = (const int*)d_in[1];        // [2, E] flat: src then dst
    const float* Wc = (const float*)d_in[2];
    const float* bc = (const float*)d_in[3];
    const float* W1 = (const float*)d_in[4];
    const float* b1 = (const float*)d_in[5];
    const float* W2 = (const float*)d_in[6];
    const float* b2 = (const float*)d_in[7];
    const float* W3 = (const float*)d_in[8];
    const float* b3 = (const float*)d_in[9];

    char* ws = (char*)d_ws;
    float*  deg = (float*)ws;                                   // 4 MB (reused as dinv)
    float*  acc = (float*)(ws + (size_t)(4 << 20));             // 128 MB
    __bf16* Bhi = (__bf16*)(ws + (size_t)(4 << 20) + 128000000ull);  // 160 KB
    __bf16* Blo = Bhi + 16 * 10 * 32 * 16;                      // 160 KB

    float* fea   = (float*)d_out;            // [100K, 256]
    float* mu    = fea + 25600000;           // [100K]
    float* theta = mu + 100000;              // [100K]

    const int* src = ei;
    const int* dst = ei + N_EDGES;

    k_init_deg<<<(N_NODES + 255) / 256, 256, 0, stream>>>(deg);
    k_pack_w1 <<<(16 * 10 * 32 * 16 + 255) / 256, 256, 0, stream>>>(W1, Bhi, Blo);
    k_deg     <<<(N_EDGES + 255) / 256, 256, 0, stream>>>(dst, deg);
    k_dinv    <<<(N_NODES + 255) / 256, 256, 0, stream>>>(deg);
    k_self    <<<(N_NODES * 32) / 256, 256, 0, stream>>>(x, Wc, deg, acc);
    k_scatter <<<(N_EDGES / 256) * 32, 256, 0, stream>>>(src, dst, x, Wc, deg, acc);
    k_gemm    <<<NGRAPH / 16, 256, 0, stream>>>(acc, bc, Bhi, Blo, b1, fea);
    k_head    <<<NGRAPH / 8, 256, 0, stream>>>(fea, W2, b2, W3, b3, mu, theta);
}
